// myRNN_9818295239328
// MI455X (gfx1250) — compile-verified, matrix operand order checked by reading, not run on hardware
//
#include <hip/hip_runtime.h>
#include <math.h>

// ---------------------------------------------------------------------------
// 2-layer tanh RNN scan for MI455X (gfx1250, wave32, WMMA).
// B=256, T=1024, I=64, H=32.  Batch is tiled M=16 per workgroup (16 WGs),
// one wave each; weights held in VGPRs as bf16 WMMA B-fragments; recurrent
// state relayout C->A goes through 2KB of LDS per layer per step.
// ---------------------------------------------------------------------------

typedef __attribute__((ext_vector_type(16))) __bf16 v16bf;
typedef __attribute__((ext_vector_type(8)))  float  v8f;

#define WMMA_BF16(A, Bm, C) \
  __builtin_amdgcn_wmma_f32_16x16x32_bf16(false, (A), false, (Bm), (short)0, (C), false, false)

constexpr int kB = 256;
constexpr int kT = 1024;
constexpr int kI = 64;
constexpr int kH = 32;

static __device__ __forceinline__ float tanh_fast(float x) {
#if __has_builtin(__builtin_amdgcn_tanhf)
  return __builtin_amdgcn_tanhf(x);   // V_TANH_F32 (gfx1250 TRANS op)
#else
  return tanhf(x);
#endif
}

// Build a bf16 B-fragment (32x16, K x N) from row-major f32 W[ldK x ld].
// Layout (mirror of the 16-bit A layout in CDNA5 ISA §7.12.2):
//   lane = n + 16*half holds B[k][n] for k in {half*8+0..7} u {16+half*8+0..7},
//   packed as (2k, 2k+1) pairs per dword slot.
static __device__ __forceinline__ v16bf load_bfrag(const float* __restrict__ W, int ld,
                                                   int k0, int n0, int ln, int half) {
  v16bf r;
#pragma unroll
  for (int blk = 0; blk < 2; ++blk) {
#pragma unroll
    for (int p = 0; p < 4; ++p) {
      const int k = k0 + blk * 16 + half * 8 + p * 2;
      r[blk * 8 + p * 2 + 0] = (__bf16)W[(k + 0) * ld + n0 + ln];
      r[blk * 8 + p * 2 + 1] = (__bf16)W[(k + 1) * ld + n0 + ln];
    }
  }
  return r;
}

// Load an A-fragment (16x32 bf16) of the hidden state from a 16x32 f32 LDS tile.
static __device__ __forceinline__ v16bf load_afrag_lds(const float* S, int ln, int half) {
  v16bf r;
#pragma unroll
  for (int blk = 0; blk < 2; ++blk) {
#pragma unroll
    for (int p = 0; p < 4; ++p) {
      const int k = blk * 16 + half * 8 + p * 2;
      r[blk * 8 + p * 2 + 0] = (__bf16)S[ln * kH + k + 0];
      r[blk * 8 + p * 2 + 1] = (__bf16)S[ln * kH + k + 1];
    }
  }
  return r;
}

// Scatter a 16x16 f32 C-fragment into a 16x32 f32 LDS tile at column n0.
// C layout: vgpr r holds row (half*8 + r), col = lane&15.
static __device__ __forceinline__ void store_cfrag_lds(float* S, int n0, const v8f c,
                                                       int ln, int half) {
#pragma unroll
  for (int r = 0; r < 8; ++r) S[(half * 8 + r) * kH + n0 + ln] = c[r];
}

static __device__ __forceinline__ v8f splat8(float v) {
  v8f c;
#pragma unroll
  for (int r = 0; r < 8; ++r) c[r] = v;
  return c;
}

// Load lane's share of the x tile for timestep t: row m=ln of the batch tile,
// four 8-float runs at cols {0,16,32,48} + half*8 (maps 1:1 onto A-fragment slots).
static __device__ __forceinline__ void load_x(const float* __restrict__ xrow, int t,
                                              int half, float4* v) {
  const float4* p = (const float4*)(xrow + (size_t)t * kI);
#pragma unroll
  for (int r = 0; r < 4; ++r) {
    const int q = r * 4 + half * 2;
    v[r * 2 + 0] = p[q + 0];
    v[r * 2 + 1] = p[q + 1];
  }
}

// Convert staged x registers into the two K-block A-fragments (cols 0-31, 32-63).
static __device__ __forceinline__ void cvt_x(const float4* v, v16bf& a0, v16bf& a1) {
#pragma unroll
  for (int r = 0; r < 4; ++r) {
    const float4 u0 = v[r * 2 + 0];
    const float4 u1 = v[r * 2 + 1];
    const int e = (r & 1) * 8;
    if (r < 2) {
      a0[e + 0] = (__bf16)u0.x; a0[e + 1] = (__bf16)u0.y;
      a0[e + 2] = (__bf16)u0.z; a0[e + 3] = (__bf16)u0.w;
      a0[e + 4] = (__bf16)u1.x; a0[e + 5] = (__bf16)u1.y;
      a0[e + 6] = (__bf16)u1.z; a0[e + 7] = (__bf16)u1.w;
    } else {
      a1[e + 0] = (__bf16)u0.x; a1[e + 1] = (__bf16)u0.y;
      a1[e + 2] = (__bf16)u0.z; a1[e + 3] = (__bf16)u0.w;
      a1[e + 4] = (__bf16)u1.x; a1[e + 5] = (__bf16)u1.y;
      a1[e + 6] = (__bf16)u1.z; a1[e + 7] = (__bf16)u1.w;
    }
  }
}

__global__ __launch_bounds__(32, 1) void rnn2_scan_kernel(
    const float* __restrict__ x,
    const float* __restrict__ Wxh1, const float* __restrict__ Whh1,
    const float* __restrict__ b1,
    const float* __restrict__ Wxh2, const float* __restrict__ Whh2,
    const float* __restrict__ b2,
    float* __restrict__ out, float* __restrict__ hid) {
  const int lane = threadIdx.x & 31;
  const int half = lane >> 4;
  const int ln   = lane & 15;
  const int b0   = blockIdx.x * 16;   // batch tile base

  __shared__ float sh1[16 * kH];      // layer-1 state relayout buffer (f32)
  __shared__ float sh2[16 * kH];      // layer-2 state relayout buffer (f32)

  // ---- one-time: weight B-fragments into VGPRs (kept live for all 1024 steps)
  v16bf WB1[2][2], WH1[2], WX2[2], WH2[2];
#pragma unroll
  for (int nt = 0; nt < 2; ++nt) {
    WB1[0][nt] = load_bfrag(Wxh1, kH, 0,  nt * 16, ln, half);
    WB1[1][nt] = load_bfrag(Wxh1, kH, 32, nt * 16, ln, half);
    WH1[nt]    = load_bfrag(Whh1, kH, 0,  nt * 16, ln, half);
    WX2[nt]    = load_bfrag(Wxh2, kH, 0,  nt * 16, ln, half);
    WH2[nt]    = load_bfrag(Whh2, kH, 0,  nt * 16, ln, half);
  }
  const float bias1v[2] = { b1[ln], b1[16 + ln] };
  const float bias2v[2] = { b2[ln], b2[16 + ln] };

  // h0 = 0 for both layers (A-fragment form)
  v16bf h1a, h2a;
#pragma unroll
  for (int e = 0; e < 16; ++e) { h1a[e] = (__bf16)0.0f; h2a[e] = (__bf16)0.0f; }

  const float* xrow = x + (size_t)(b0 + ln) * kT * kI;  // this lane's batch row
  float4 xv[8];
  load_x(xrow, 0, half, xv);                            // prologue: t=0

#pragma unroll 2
  for (int t = 0; t < kT; ++t) {
    v16bf xa0, xa1;
    cvt_x(xv, xa0, xa1);
    if (t + 1 < kT) load_x(xrow, t + 1, half, xv);      // register double-buffer
    if (t + 2 < kT) __builtin_prefetch(xrow + (size_t)(t + 2) * kI, 0, 0);

    // -------- layer 1: h1 = tanh(x_t@Wxh1 + b1 + h1@Whh1) --------
#pragma unroll
    for (int nt = 0; nt < 2; ++nt) {
      v8f c = splat8(bias1v[nt]);
      c = WMMA_BF16(xa0, WB1[0][nt], c);   // x cols 0..31
      c = WMMA_BF16(xa1, WB1[1][nt], c);   // x cols 32..63
      c = WMMA_BF16(h1a, WH1[nt], c);      // recurrent term (critical chain)
#pragma unroll
      for (int r = 0; r < 8; ++r) c[r] = tanh_fast(c[r]);
      store_cfrag_lds(sh1, nt * 16, c, ln, half);
    }
    h1a = load_afrag_lds(sh1, ln, half);   // C-layout -> A-layout (in-order DS)

    // -------- layer 2: h2 = tanh(h1@Wxh2 + b2 + h2@Whh2) --------
#pragma unroll
    for (int nt = 0; nt < 2; ++nt) {
      v8f c = splat8(bias2v[nt]);
      c = WMMA_BF16(h1a, WX2[nt], c);
      c = WMMA_BF16(h2a, WH2[nt], c);
#pragma unroll
      for (int r = 0; r < 8; ++r) c[r] = tanh_fast(c[r]);
      store_cfrag_lds(sh2, nt * 16, c, ln, half);
      // emit outputs (off the critical chain, fire-and-forget stores)
#pragma unroll
      for (int r = 0; r < 8; ++r) {
        const int m = half * 8 + r;
        out[((size_t)(b0 + m) * kT + t) * kH + nt * 16 + ln] = c[r];
      }
    }
    h2a = load_afrag_lds(sh2, ln, half);
  }

  // -------- final hidden states: concat(h1, h2) along features --------
  for (int idx = lane; idx < 16 * 64; idx += 32) {
    const int m = idx >> 6;
    const int c = idx & 63;
    hid[(size_t)(b0 + m) * 64 + c] = (c < kH) ? sh1[m * kH + c] : sh2[m * kH + (c - kH)];
  }
}

extern "C" void kernel_launch(void* const* d_in, const int* in_sizes, int n_in,
                              void* d_out, int out_size, void* d_ws, size_t ws_size,
                              hipStream_t stream) {
  (void)in_sizes; (void)n_in; (void)d_ws; (void)ws_size; (void)out_size;
  const float* x    = (const float*)d_in[0];
  const float* Wxh1 = (const float*)d_in[1];
  const float* Whh1 = (const float*)d_in[2];
  const float* b1   = (const float*)d_in[3];
  const float* Wxh2 = (const float*)d_in[4];
  const float* Whh2 = (const float*)d_in[5];
  const float* b2   = (const float*)d_in[6];

  float* out = (float*)d_out;                         // [B, T, H]
  float* hid = out + (size_t)kB * kT * kH;            // [B, 2H]

  rnn2_scan_kernel<<<kB / 16, 32, 0, stream>>>(x, Wxh1, Whh1, b1, Wxh2, Whh2, b2, out, hid);
}
